// TernaryDense_17892833755667
// MI455X (gfx1250) — matrix operand order repaired and checked by reading, not compiled, not run on hardware
//
#include <hip/hip_runtime.h>
#include <hip/hip_bf16.h>
#include <stddef.h>

// ---------------------------------------------------------------------------
// TernaryDense: out[8192,2048] = alpha * (x @ t) + b,
//   alpha = mean(|w|), t = ternary(w) in {-1,0,+1}  (exact in f16)
// GEMM runs on v_wmma_f32_16x16x32_f16 (CDNA5 matrix path, wave32).
// ---------------------------------------------------------------------------

#define M_DIM 8192
#define K_DIM 2048
#define N_DIM 2048

typedef __attribute__((ext_vector_type(16))) _Float16 v16h;
typedef __attribute__((ext_vector_type(8)))  _Float16 v8h;
typedef __attribute__((ext_vector_type(8)))  float    v8f;

union V16 { v16h v; v8h h[2]; };

// ---------------------------------------------------------------- pass 0: zero
__global__ void zero_alpha_kernel(float* alpha_sum) {
    if (threadIdx.x == 0 && blockIdx.x == 0) alpha_sum[0] = 0.0f;
}

// ------------------------------------------------------- pass 1: sum(|w|) f32
// 4M elements, 4 per thread (float4), wave32 shuffle reduce, 1 atomic / wave.
__global__ __launch_bounds__(256) void absum_kernel(const float* __restrict__ w,
                                                    float* __restrict__ alpha_sum) {
    size_t i = ((size_t)blockIdx.x * blockDim.x + threadIdx.x) * 4;
    float4 v = *(const float4*)(w + i);
    float s = fabsf(v.x) + fabsf(v.y) + fabsf(v.z) + fabsf(v.w);
#pragma unroll
    for (int o = 16; o > 0; o >>= 1) s += __shfl_down(s, o, 32);
    if ((threadIdx.x & 31) == 0) atomicAdd(alpha_sum, s);
}

// -------------------------------- pass 2: quantize + transpose w -> t[n][k] f16
// w is [K][N] row-major; store t as [N][K] so B-fragment reads are K-contiguous.
__global__ __launch_bounds__(256) void quant_tr_kernel(const float* __restrict__ w,
                                                       _Float16* __restrict__ tB,
                                                       const float* __restrict__ alpha_sum) {
    __shared__ _Float16 tile[32][33];
    const float alpha = alpha_sum[0] * (1.0f / ((float)K_DIM * (float)N_DIM));
    const float thr = 0.7f * alpha;
    const int kbase = blockIdx.y * 32;
    const int nbase = blockIdx.x * 32;
    const int tx = threadIdx.x;   // 0..31
    const int ty = threadIdx.y;   // 0..7
#pragma unroll
    for (int j = 0; j < 32; j += 8) {
        float v = w[(size_t)(kbase + ty + j) * N_DIM + nbase + tx];
        float q = (fabsf(v) < thr) ? 0.0f : (v < 0.0f ? -1.0f : 1.0f);
        tile[ty + j][tx] = (_Float16)q;
    }
    __syncthreads();
#pragma unroll
    for (int j = 0; j < 32; j += 8) {
        tB[(size_t)(nbase + ty + j) * K_DIM + kbase + tx] = tile[tx][ty + j];
    }
}

// ---------------------------------------------------- pass 3: x f32 -> f16
__global__ __launch_bounds__(256) void cvt_x_kernel(const float* __restrict__ x,
                                                    _Float16* __restrict__ xh) {
    size_t i = ((size_t)blockIdx.x * blockDim.x + threadIdx.x) * 8;
    float4 a = *(const float4*)(x + i);
    float4 c = *(const float4*)(x + i + 4);
    v8h o;
    o[0] = (_Float16)a.x; o[1] = (_Float16)a.y; o[2] = (_Float16)a.z; o[3] = (_Float16)a.w;
    o[4] = (_Float16)c.x; o[5] = (_Float16)c.y; o[6] = (_Float16)c.z; o[7] = (_Float16)c.w;
    *(v8h*)(xh + i) = o;
}

// ------------------------------------------------------------- pass 4: GEMM
// Block tile 128x128, BK=32; 256 threads = 8 waves in a 4(M) x 2(N) grid;
// each wave computes 32x64 = 2x4 WMMA tiles of 16x16 with f32 accumulators.
#define BM 128
#define BN 128
#define BK 32
#define LDP 40   // LDS row pitch in halves: 80B = 20 banks -> conflict-free

__global__ __launch_bounds__(256) void gemm_wmma_kernel(const _Float16* __restrict__ A,   // [M][K] f16
                                                        const _Float16* __restrict__ B,   // [N][K] f16 (t transposed)
                                                        const float* __restrict__ bias,   // [N]
                                                        const float* __restrict__ alpha_sum,
                                                        float* __restrict__ out) {        // [M][N]
    __shared__ _Float16 As[BM][LDP];
    __shared__ _Float16 Bs[BN][LDP];

    const int tid  = threadIdx.x;
    const int lane = tid & 31;
    const int wid  = tid >> 5;
    const int wm   = wid & 3;          // wave M index 0..3  -> 32 rows each
    const int wn   = wid >> 2;         // wave N index 0..1  -> 64 cols each
    const int mblk = blockIdx.y * BM;
    const int nblk = blockIdx.x * BN;

    const int rl = lane & 15;          // row/col within 16
    const int ch = lane >> 4;          // lane-half selector (ISA K grouping)

    v8f acc[2][4] = {};

    for (int kt = 0; kt < K_DIM / BK; ++kt) {
        const int kbase = kt * BK;
        // Cooperative tile loads: 128 rows x 64B = 512 x 16B chunks, 2 per thread.
#pragma unroll
        for (int i = 0; i < 2; ++i) {
            int idx = tid + i * 256;
            int row = idx >> 2;
            int c16 = idx & 3;
            *(v8h*)&As[row][c16 * 8] =
                *(const v8h*)(A + (size_t)(mblk + row) * K_DIM + kbase + c16 * 8);
        }
#pragma unroll
        for (int i = 0; i < 2; ++i) {
            int idx = tid + i * 256;
            int col = idx >> 2;
            int c16 = idx & 3;
            *(v8h*)&Bs[col][c16 * 8] =
                *(const v8h*)(B + (size_t)(nblk + col) * K_DIM + kbase + c16 * 8);
        }
        __syncthreads();

        // A fragment (16x32 f16): lane<16 holds K0..7 & K16..23 of row rl;
        // lane>=16 holds K8..15 & K24..31 (per ISA 16-bit A layout).
        V16 afrag[2];
#pragma unroll
        for (int mi = 0; mi < 2; ++mi) {
            int row = wm * 32 + mi * 16 + rl;
            afrag[mi].h[0] = *(const v8h*)&As[row][ch * 8];
            afrag[mi].h[1] = *(const v8h*)&As[row][16 + ch * 8];
        }
        // B fragment (32x16 f16): lane = column; lane<16 holds K0..7 & K16..23, lane>=16 holds K8..15 & K24..31.
        V16 bfrag[4];
#pragma unroll
        for (int ni = 0; ni < 4; ++ni) {
            int col = wn * 64 + ni * 16 + rl;
            bfrag[ni].h[0] = *(const v8h*)&Bs[col][ch * 8];
            bfrag[ni].h[1] = *(const v8h*)&Bs[col][16 + ch * 8];
        }

#pragma unroll
        for (int mi = 0; mi < 2; ++mi)
#pragma unroll
            for (int ni = 0; ni < 4; ++ni)
                acc[mi][ni] = __builtin_amdgcn_wmma_f32_16x16x32_f16(
                    /*neg_a=*/false, afrag[mi].v,
                    /*neg_b=*/false, bfrag[ni].v,
                    /*c_mod=*/(short)0, acc[mi][ni],
                    /*reuse_a=*/false, /*reuse_b=*/false);
        __syncthreads();
    }

    // Epilogue: out = alpha*acc + b.  C/D layout: lane%16 = N, lane/16 -> M+8,
    // VGPR r -> M row r (r+8 for upper half-wave).
    const float alpha = alpha_sum[0] * (1.0f / ((float)K_DIM * (float)N_DIM));
    const int rh = ch * 8;
#pragma unroll
    for (int mi = 0; mi < 2; ++mi) {
#pragma unroll
        for (int ni = 0; ni < 4; ++ni) {
            int n0 = nblk + wn * 64 + ni * 16 + rl;
            int m0 = mblk + wm * 32 + mi * 16 + rh;
            float bv = bias[n0];
#pragma unroll
            for (int r = 0; r < 8; ++r) {
                out[(size_t)(m0 + r) * N_DIM + n0] = alpha * acc[mi][ni][r] + bv;
            }
        }
    }
}

// ---------------------------------------------------------------------------
extern "C" void kernel_launch(void* const* d_in, const int* in_sizes, int n_in,
                              void* d_out, int out_size, void* d_ws, size_t ws_size,
                              hipStream_t stream) {
    const float* x = (const float*)d_in[0];   // [8192, 2048] f32
    const float* w = (const float*)d_in[1];   // [2048, 2048] f32
    const float* b = (const float*)d_in[2];   // [2048] f32
    float* out = (float*)d_out;               // [8192, 2048] f32

    // Workspace layout
    float*    alpha_sum = (float*)d_ws;
    _Float16* xh = (_Float16*)((char*)d_ws + 256);
    _Float16* tB = (_Float16*)((char*)d_ws + 256 + (size_t)M_DIM * K_DIM * sizeof(_Float16));

    // 0) zero alpha accumulator
    zero_alpha_kernel<<<1, 32, 0, stream>>>(alpha_sum);

    // 1) sum(|w|): 4M elems, 4/thread -> 1M threads
    {
        int threads = 256;
        int blocks = (K_DIM * N_DIM / 4) / threads;  // 4096
        absum_kernel<<<blocks, threads, 0, stream>>>(w, alpha_sum);
    }

    // 2) quantize + transpose w -> tB [N][K] f16
    {
        dim3 grid(N_DIM / 32, K_DIM / 32);  // (64, 64)
        dim3 block(32, 8);
        quant_tr_kernel<<<grid, block, 0, stream>>>(w, tB, alpha_sum);
    }

    // 3) x f32 -> f16
    {
        int threads = 256;
        int blocks = (int)(((size_t)M_DIM * K_DIM / 8) / threads);  // 8192
        cvt_x_kernel<<<blocks, threads, 0, stream>>>(x, xh);
    }

    // 4) WMMA GEMM + epilogue
    {
        dim3 grid(N_DIM / BN, M_DIM / BM);  // (16, 64)
        gemm_wmma_kernel<<<grid, 256, 0, stream>>>(xh, tB, b, alpha_sum, out);
    }
}
